// Transformer_48825188221151
// MI455X (gfx1250) — matrix operand order repaired and checked by reading, not compiled, not run on hardware
//
#include <hip/hip_runtime.h>
#include <hip/hip_bf16.h>

// ---------------------------------------------------------------------------
// Transformer encoder forward for MI455X (gfx1250), wave32 + WMMA f16 + TDM.
// B=8, T=512, D=512, H=8, DQ=DV=64, L=6, D_HID=2048, classes=5.
// All GEMM operands are staged by the Tensor Data Mover: weights are
// pre-transposed to [N,K] once per launch, V is produced head-transposed,
// so every B tile is a straight 2D copy the TDM can move.
// ---------------------------------------------------------------------------

typedef __attribute__((ext_vector_type(16))) _Float16 v16h;
typedef __attribute__((ext_vector_type(8)))  _Float16 v8h;
typedef __attribute__((ext_vector_type(8)))  float    v8f;

union F16x16 { v16h v; v8h h[2]; };

#define LDS_STRIDE 40   // 32 data + 8 pad halfs: 16B-aligned rows, conflict-free

// ---- Tensor Data Mover availability ---------------------------------------
#if __has_builtin(__builtin_amdgcn_tensor_load_to_lds) && \
    __has_builtin(__builtin_amdgcn_s_wait_tensorcnt)
  #define USE_TDM 1
  #if __has_include(<hip/amd_detail/amd_gfx1250_TDM.h>)
    #define TDM_6ARG 1     // therock-10.0 headers => clang-23 6-arg builtin
  #else
    #define TDM_6ARG 0     // ROCm 7.2 clang-22 5-arg builtin
  #endif
#else
  #define USE_TDM 0
#endif

#if USE_TDM
typedef unsigned u32x4 __attribute__((ext_vector_type(4)));
typedef int      i32x4 __attribute__((ext_vector_type(4)));
typedef int      i32x8 __attribute__((ext_vector_type(8)));

// LDS aperture maps addr[31:0] -> LDS byte offset, so truncation is exact.
__device__ __forceinline__ unsigned lds_off(const void* p) {
    return (unsigned)(unsigned long long)p;
}

// 2D tile DMA: tileK contiguous f16 elems per row, tileRows rows, row stride
// strideElems (f16). LDS destination padded to LDS_STRIDE halfs per row via
// the descriptor pad fields (interval = 16 DWORDs, amount = 4 DWORDs).
__device__ __forceinline__ void tdm_load_2d(unsigned ldsByte, const void* gp,
                                            unsigned tileK, unsigned tileRows,
                                            long strideElems) {
    unsigned long long ga = (unsigned long long)gp;
    u32x4 g0;
    g0.x = 1u;                                             // count=1, user mode
    g0.y = ldsByte;                                        // lds_addr
    g0.z = (unsigned)ga;                                   // global_addr lo
    g0.w = (unsigned)((ga >> 32) & 0x01FFFFFFu) | (2u << 30); // ga hi | type=2
    unsigned long long st = (unsigned long long)strideElems;
    i32x8 g1;
    g1[0] = (int)((1u << 16) |            // data_size = 2 bytes
                  (1u << 20) |            // pad_enable
                  (3u << 22) |            // pad_interval: 16 DWORDs
                  (3u << 25));            // pad_amount: 4 DWORDs
    g1[1] = (int)((tileK & 0xFFFFu) << 16);                   // tensor_dim0 lo16
    g1[2] = (int)((tileK >> 16) | ((tileRows & 0xFFFFu) << 16)); // dim0 hi | dim1 lo
    g1[3] = (int)((tileRows >> 16) | (tileK << 16));          // dim1 hi | tile_dim0
    g1[4] = (int)tileRows;                                    // tile_dim1 (dim2=0)
    g1[5] = (int)(unsigned)st;                                // dim0_stride lo32
    g1[6] = (int)(unsigned)((st >> 32) & 0xFFFFu);            // dim0_stride hi16
    g1[7] = 0;
    i32x4 z4 = {0, 0, 0, 0};
#if TDM_6ARG
    i32x8 z8 = {0, 0, 0, 0, 0, 0, 0, 0};
    __builtin_amdgcn_tensor_load_to_lds(g0, g1, z4, z4, z8, 0);
#else
    __builtin_amdgcn_tensor_load_to_lds(g0, g1, z4, z4, 0);
#endif
}
#endif  // USE_TDM

// ---------------------------------------------------------------------------
// Weight transpose + f32->f16: src[l][k][n] -> dst[l][n][k].  32x32 LDS tiles.
// ---------------------------------------------------------------------------
__global__ __launch_bounds__(256)
void transpose_w_kernel(const float* __restrict__ src, _Float16* __restrict__ dst,
                        int K, int N) {
    __shared__ _Float16 tile[32][33];
    const long slab = (long)blockIdx.z * K * N;
    const int k0 = blockIdx.y * 32, n0 = blockIdx.x * 32;
    const int tx = threadIdx.x & 31, ty = threadIdx.x >> 5;  // 32 x 8
#pragma unroll
    for (int j = 0; j < 32; j += 8)
        tile[ty + j][tx] = (_Float16)src[slab + (long)(k0 + ty + j) * N + n0 + tx];
    __syncthreads();
#pragma unroll
    for (int j = 0; j < 32; j += 8)
        dst[slab + (long)(n0 + ty + j) * K + k0 + tx] = tile[tx][ty + j];
}

// ---------------------------------------------------------------------------
// Embedding + sinusoidal positional encoding. One block per (b,t) row.
// ---------------------------------------------------------------------------
__global__ __launch_bounds__(128)
void embed_pe_kernel(const int* __restrict__ x, const float* __restrict__ emb,
                     float* __restrict__ hf, _Float16* __restrict__ hh) {
    const long row = blockIdx.x;          // b*T + t
    const int  t   = (int)(row & 511);    // T = 512
    const long tok = (long)x[row];
    const float kLog = 0.017988946039347f; // ln(10000)/512
#pragma unroll
    for (int i = 0; i < 4; ++i) {
        int c = threadIdx.x + i * 128;
        float inv = __expf(-(float)(c & ~1) * kLog);
        float ang = (float)t * inv;
        float pe  = (c & 1) ? __cosf(ang) : __sinf(ang);
        float v   = emb[tok * 512 + c] + pe;
        hf[row * 512 + c] = v;
        hh[row * 512 + c] = (_Float16)v;
    }
}

// ---------------------------------------------------------------------------
// Tiled WMMA GEMM: C[M,N] = A[M,K] @ B^T[N,K] + bias (both f16, f32 accum).
// 64x64 block tile, 128 threads (4 waves, each 32x32 = 2x2 WMMA fragments).
// Double-buffered LDS; both tiles staged by the Tensor Data Mover from
// wave 0, prefetched one K-step ahead (s_wait_tensorcnt pipelining).
// CT=true: write C as f16 head-transposed vT[b][n][t] (V projection).
// ---------------------------------------------------------------------------
template <bool RELU, bool CT>
__global__ __launch_bounds__(128)
void gemm64_wmma(const _Float16* __restrict__ A, const _Float16* __restrict__ B,
                 const float* __restrict__ bias,
                 float* __restrict__ Cf, _Float16* __restrict__ Ch,
                 int lda, int ldb, int ldc, int K, int H,
                 long aB, long aH, long bB, long bH, long cB, long cH) {
    __shared__ _Float16 sA[2][64 * LDS_STRIDE];
    __shared__ _Float16 sB[2][64 * LDS_STRIDE];

    const int tid = threadIdx.x;
    const int z  = blockIdx.z;
    const int zb = z / H, zh = z % H;
    A += zb * aB + zh * aH;
    B += zb * bB + zh * bH;
    const long coff = CT ? 0 : (zb * cB + zh * cH);

    const int bm = blockIdx.y * 64;
    const int bn = blockIdx.x * 64;

    const int wave = tid >> 5;
    const int lane = tid & 31;
    const int wm   = (wave >> 1) * 32;
    const int wn   = (wave & 1) * 32;
    const int half = lane >> 4;
    const int l16  = lane & 15;

    const int nT = K >> 5;
    v8f acc[2][2] = {};

#if USE_TDM
    auto tdmIssue = [&](int t, int c) {
        tdm_load_2d(lds_off(&sA[c][0]), A + (long)bm * lda + (long)t * 32,
                    32u, 64u, (long)lda);
        tdm_load_2d(lds_off(&sB[c][0]), B + (long)bn * ldb + (long)t * 32,
                    32u, 64u, (long)ldb);
    };
    if (wave == 0) tdmIssue(0, 0);       // DMA tile 0
#else
    // manual fallback: register-staged straight copies
    const int ar = tid >> 2, ac = (tid & 3) * 8;   // rows ar, ar+32
    const _Float16* aRow0 = A + (long)(bm + ar) * lda + ac;
    const _Float16* aRow1 = A + (long)(bm + ar + 32) * lda + ac;
    const _Float16* bRow0 = B + (long)(bn + ar) * ldb + ac;
    const _Float16* bRow1 = B + (long)(bn + ar + 32) * ldb + ac;
    v8h rA0, rA1, rB0, rB1;
    auto loadRegs = [&](int t) {
        rA0 = *(const v8h*)(aRow0 + (long)t * 32);
        rA1 = *(const v8h*)(aRow1 + (long)t * 32);
        rB0 = *(const v8h*)(bRow0 + (long)t * 32);
        rB1 = *(const v8h*)(bRow1 + (long)t * 32);
    };
    auto storeRegs = [&](int c) {
        *(v8h*)&sA[c][ar * LDS_STRIDE + ac]        = rA0;
        *(v8h*)&sA[c][(ar + 32) * LDS_STRIDE + ac] = rA1;
        *(v8h*)&sB[c][ar * LDS_STRIDE + ac]        = rB0;
        *(v8h*)&sB[c][(ar + 32) * LDS_STRIDE + ac] = rB1;
    };
    loadRegs(0);
    storeRegs(0);
#endif

    for (int i = 0; i < nT; ++i) {
        const int ci = i & 1;
#if USE_TDM
        if (wave == 0) {                 // per-wave DMA engine, wave 0 drives it
            if (i + 1 < nT) {
                tdmIssue(i + 1, ci ^ 1); // prefetch next tile into other buffer
                __builtin_amdgcn_s_wait_tensorcnt((short)2);
            } else {
                __builtin_amdgcn_s_wait_tensorcnt((short)0);
            }
        }
#endif
        __syncthreads();                 // tile i buffers ready for everyone
#if !USE_TDM
        if (i + 1 < nT) loadRegs(i + 1); // global loads overlap the math below
#endif
        // ---- fragment loads (layouts per CDNA5 ISA 7.12.2) -----------------
        F16x16 af[2], bf[2];
#pragma unroll
        for (int mi = 0; mi < 2; ++mi) {
            int r = wm + mi * 16 + l16;
            af[mi].h[0] = *(const v8h*)&sA[ci][r * LDS_STRIDE + half * 8];
            af[mi].h[1] = *(const v8h*)&sA[ci][r * LDS_STRIDE + 16 + half * 8];
        }
#pragma unroll
        for (int ni = 0; ni < 2; ++ni) {
            int n = wn + ni * 16 + l16;
            bf[ni].h[0] = *(const v8h*)&sB[ci][n * LDS_STRIDE + half * 8];
            bf[ni].h[1] = *(const v8h*)&sB[ci][n * LDS_STRIDE + 16 + half * 8];
        }
#pragma unroll
        for (int mi = 0; mi < 2; ++mi)
#pragma unroll
            for (int ni = 0; ni < 2; ++ni)
                acc[mi][ni] = __builtin_amdgcn_wmma_f32_16x16x32_f16(
                    false, af[mi].v, false, bf[ni].v,
                    (short)0, acc[mi][ni], false, false);
        __syncthreads();                 // everyone done reading tile i buffers
#if !USE_TDM
        if (i + 1 < nT) storeRegs(ci ^ 1);
#endif
    }

    // ---- epilogue ----------------------------------------------------------
#pragma unroll
    for (int mi = 0; mi < 2; ++mi) {
#pragma unroll
        for (int ni = 0; ni < 2; ++ni) {
            int n = bn + wn + ni * 16 + l16;
            float bv = bias ? bias[n] : 0.0f;
            if (CT) {
                // head-transposed f16 store: vT[b][n][t], 8 consecutive t
                int m0 = bm + wm + mi * 16 + half * 8;
                v8h hv;
#pragma unroll
                for (int r = 0; r < 8; ++r) {
                    float v = acc[mi][ni][r] + bv;
                    if (RELU) v = v > 0.0f ? v : 0.0f;
                    hv[r] = (_Float16)v;
                }
                long base = (long)(m0 >> 9) * cB + (long)n * ldc + (m0 & 511);
                *(v8h*)&Ch[base] = hv;
            } else {
#pragma unroll
                for (int r = 0; r < 8; ++r) {
                    int m = bm + wm + mi * 16 + half * 8 + r;
                    float v = acc[mi][ni][r] + bv;
                    if (RELU) v = v > 0.0f ? v : 0.0f;
                    long idx = coff + (long)m * ldc + n;
                    if (Cf) Cf[idx] = v;
                    if (Ch) Ch[idx] = (_Float16)v;
                }
            }
        }
    }
}

// ---------------------------------------------------------------------------
// Row softmax over 512 (no 1/sqrt(d) scaling, per reference). One block/row.
// ---------------------------------------------------------------------------
__global__ __launch_bounds__(128)
void softmax512_kernel(const float* __restrict__ S, _Float16* __restrict__ P) {
    __shared__ float red[128];
    const long row = blockIdx.x;
    const float* s = S + row * 512;
    float v[4];
    float m = -1e30f;
#pragma unroll
    for (int i = 0; i < 4; ++i) { v[i] = s[threadIdx.x + i * 128]; m = fmaxf(m, v[i]); }
    red[threadIdx.x] = m; __syncthreads();
    for (int off = 64; off > 0; off >>= 1) {
        if (threadIdx.x < off) red[threadIdx.x] = fmaxf(red[threadIdx.x], red[threadIdx.x + off]);
        __syncthreads();
    }
    m = red[0]; __syncthreads();
    float sum = 0.0f;
#pragma unroll
    for (int i = 0; i < 4; ++i) { v[i] = __expf(v[i] - m); sum += v[i]; }
    red[threadIdx.x] = sum; __syncthreads();
    for (int off = 64; off > 0; off >>= 1) {
        if (threadIdx.x < off) red[threadIdx.x] += red[threadIdx.x + off];
        __syncthreads();
    }
    float inv = 1.0f / red[0];
    _Float16* p = P + row * 512;
#pragma unroll
    for (int i = 0; i < 4; ++i) p[threadIdx.x + i * 128] = (_Float16)(v[i] * inv);
}

// ---------------------------------------------------------------------------
// y = LayerNorm(a + res) with gamma/beta; writes f32 (next residual) and f16.
// ---------------------------------------------------------------------------
__global__ __launch_bounds__(128)
void add_ln512_kernel(const float* __restrict__ a, const float* __restrict__ res,
                      const float* __restrict__ g, const float* __restrict__ bta,
                      float* __restrict__ hf, _Float16* __restrict__ hh) {
    __shared__ float red[128];
    const long row = blockIdx.x;
    float v[4];
    float sum = 0.0f;
#pragma unroll
    for (int i = 0; i < 4; ++i) {
        int c = threadIdx.x + i * 128;
        v[i] = a[row * 512 + c] + res[row * 512 + c];
        sum += v[i];
    }
    red[threadIdx.x] = sum; __syncthreads();
    for (int off = 64; off > 0; off >>= 1) {
        if (threadIdx.x < off) red[threadIdx.x] += red[threadIdx.x + off];
        __syncthreads();
    }
    float mu = red[0] * (1.0f / 512.0f); __syncthreads();
    float sq = 0.0f;
#pragma unroll
    for (int i = 0; i < 4; ++i) { float d = v[i] - mu; sq += d * d; }
    red[threadIdx.x] = sq; __syncthreads();
    for (int off = 64; off > 0; off >>= 1) {
        if (threadIdx.x < off) red[threadIdx.x] += red[threadIdx.x + off];
        __syncthreads();
    }
    float rstd = rsqrtf(red[0] * (1.0f / 512.0f) + 1e-5f);
#pragma unroll
    for (int i = 0; i < 4; ++i) {
        int c = threadIdx.x + i * 128;
        float y = g[c] * (v[i] - mu) * rstd + bta[c];
        hf[row * 512 + c] = y;
        hh[row * 512 + c] = (_Float16)y;
    }
}

// ---------------------------------------------------------------------------
// out[b,c] = h[b, :] . Wout[:, c] + bout[c]; 40 blocks, fp32 reduction.
// ---------------------------------------------------------------------------
__global__ __launch_bounds__(256)
void out_head_kernel(const float* __restrict__ h, const float* __restrict__ Wout,
                     const float* __restrict__ bout, float* __restrict__ out) {
    __shared__ float red[256];
    const int b = blockIdx.x / 5;
    const int c = blockIdx.x % 5;
    const float* hb = h + (long)b * 262144;
    float sum = 0.0f;
    for (long j = threadIdx.x; j < 262144; j += 256)
        sum += hb[j] * Wout[j * 5 + c];
    red[threadIdx.x] = sum; __syncthreads();
    for (int off = 128; off > 0; off >>= 1) {
        if (threadIdx.x < off) red[threadIdx.x] += red[threadIdx.x + off];
        __syncthreads();
    }
    if (threadIdx.x == 0) out[blockIdx.x] = red[0] + bout[c];
}

// ---------------------------------------------------------------------------
// Host launch
// ---------------------------------------------------------------------------
extern "C" void kernel_launch(void* const* d_in, const int* in_sizes, int n_in,
                              void* d_out, int out_size, void* d_ws, size_t ws_size,
                              hipStream_t stream) {
    (void)in_sizes; (void)n_in; (void)out_size; (void)ws_size;
    const int*   x     = (const int*)d_in[0];
    const float* emb   = (const float*)d_in[2];
    const float* Wq    = (const float*)d_in[3];
    const float* bq    = (const float*)d_in[4];
    const float* Wk    = (const float*)d_in[5];
    const float* bk    = (const float*)d_in[6];
    const float* Wv    = (const float*)d_in[7];
    const float* bv    = (const float*)d_in[8];
    const float* Wo    = (const float*)d_in[9];
    const float* bo    = (const float*)d_in[10];
    const float* gamma = (const float*)d_in[11];
    const float* beta  = (const float*)d_in[12];
    const float* W1    = (const float*)d_in[13];
    const float* b1    = (const float*)d_in[14];
    const float* W2    = (const float*)d_in[15];
    const float* b2    = (const float*)d_in[16];
    const float* Wout  = (const float*)d_in[17];
    const float* bout  = (const float*)d_in[18];
    float* out = (float*)d_out;

    const long NL = 6, D = 512, DH = 2048, T = 512, Bb = 8, Hh = 8;
    const long M = Bb * T;                    // 4096 token rows

    // ---- workspace carve-out (256B aligned) --------------------------------
    size_t off = 0;
    auto alloc = [&](size_t bytes) -> char* {
        char* p = (char*)d_ws + off;
        off += (bytes + 255) & ~(size_t)255;
        return p;
    };
    _Float16* wqT   = (_Float16*)alloc(NL * D * D * 2);   // [L][N=512][K=512]
    _Float16* wkT   = (_Float16*)alloc(NL * D * D * 2);
    _Float16* wvT   = (_Float16*)alloc(NL * D * D * 2);
    _Float16* woT   = (_Float16*)alloc(NL * D * D * 2);
    _Float16* w1T   = (_Float16*)alloc(NL * D * DH * 2);  // [L][2048][512]
    _Float16* w2T   = (_Float16*)alloc(NL * DH * D * 2);  // [L][512][2048]
    float*    h32   = (float*)   alloc(M * D * 4);
    _Float16* h16   = (_Float16*)alloc(M * D * 2);
    _Float16* q16   = (_Float16*)alloc(M * D * 2);
    _Float16* k16   = (_Float16*)alloc(M * D * 2);
    _Float16* vT16  = (_Float16*)alloc(M * D * 2);        // [B][512][T]
    float*    S32   = (float*)   alloc(Bb * Hh * T * T * 4);
    _Float16* P16   = (_Float16*)alloc(Bb * Hh * T * T * 2);
    _Float16* o16   = (_Float16*)alloc(M * D * 2);
    float*    t32   = (float*)   alloc(M * D * 4);
    _Float16* hid16 = (_Float16*)alloc(M * DH * 2);

    // ---- transpose+convert weights once per launch -------------------------
    transpose_w_kernel<<<dim3(16, 16, 6), 256, 0, stream>>>(Wq, wqT, 512, 512);
    transpose_w_kernel<<<dim3(16, 16, 6), 256, 0, stream>>>(Wk, wkT, 512, 512);
    transpose_w_kernel<<<dim3(16, 16, 6), 256, 0, stream>>>(Wv, wvT, 512, 512);
    transpose_w_kernel<<<dim3(16, 16, 6), 256, 0, stream>>>(Wo, woT, 512, 512);
    transpose_w_kernel<<<dim3(64, 16, 6), 256, 0, stream>>>(W1, w1T, 512, 2048);
    transpose_w_kernel<<<dim3(16, 64, 6), 256, 0, stream>>>(W2, w2T, 2048, 512);

    // ---- embedding + positional encoding -----------------------------------
    embed_pe_kernel<<<M, 128, 0, stream>>>(x, emb, h32, h16);

    const dim3 gProj(D / 64, M / 64, 1);      // (8, 64)
    const dim3 gScore(T / 64, T / 64, Bb * Hh);
    const dim3 gAV(1, T / 64, Bb * Hh);
    const dim3 gF1(DH / 64, M / 64, 1);       // (32, 64)
    const dim3 gF2(D / 64, M / 64, 1);

    for (int L = 0; L < NL; ++L) {
        const _Float16* wq = wqT + (long)L * D * D;
        const _Float16* wk = wkT + (long)L * D * D;
        const _Float16* wv = wvT + (long)L * D * D;
        const _Float16* wo = woT + (long)L * D * D;
        const _Float16* w1 = w1T + (long)L * D * DH;
        const _Float16* w2 = w2T + (long)L * DH * D;
        const float* bq_ = bq + L * D;
        const float* bk_ = bk + L * D;
        const float* bv_ = bv + L * D;
        const float* bo_ = bo + L * D;
        const float* b1_ = b1 + L * DH;
        const float* b2_ = b2 + L * D;
        const float* g_  = gamma + L * D;
        const float* bt_ = beta + L * D;

        // Q, K projections: [4096,512] = h16 @ W + b -> f16
        gemm64_wmma<false, false><<<gProj, 128, 0, stream>>>(
            h16, wq, bq_, nullptr, q16, 512, 512, 512, 512, 1, 0, 0, 0, 0, 0, 0);
        gemm64_wmma<false, false><<<gProj, 128, 0, stream>>>(
            h16, wk, bk_, nullptr, k16, 512, 512, 512, 512, 1, 0, 0, 0, 0, 0, 0);
        // V projection -> head-transposed f16 vT[b][n][t]
        gemm64_wmma<false, true><<<gProj, 128, 0, stream>>>(
            h16, wv, bv_, nullptr, vT16, 512, 512, /*t-stride*/ 512, 512, 1,
            0, 0, 0, 0, /*per-b stride*/ D * T, 0);

        // scores[z,t,s] = Q_bh @ K_bh^T   (K=64, batched over b,h)
        gemm64_wmma<false, false><<<gScore, 128, 0, stream>>>(
            q16, k16, nullptr, S32, nullptr, 512, 512, 512, 64, 8,
            /*aB*/ T * 512, /*aH*/ 64, /*bB*/ T * 512, /*bH*/ 64,
            /*cB*/ Hh * T * T, /*cH*/ T * T);

        softmax512_kernel<<<Bb * Hh * T, 128, 0, stream>>>(S32, P16);

        // context o_bh = P_bh @ V_bh  (B = vT, K=512, N=64) -> o16 [4096,512]
        gemm64_wmma<false, false><<<gAV, 128, 0, stream>>>(
            P16, vT16, nullptr, nullptr, o16, 512, 512, 512, 512, 8,
            /*aB*/ Hh * T * T, /*aH*/ T * T, /*bB*/ D * T, /*bH*/ 64 * T,
            /*cB*/ T * 512, /*cH*/ 64);

        // output projection -> f32
        gemm64_wmma<false, false><<<gProj, 128, 0, stream>>>(
            o16, wo, bo_, t32, nullptr, 512, 512, 512, 512, 1, 0, 0, 0, 0, 0, 0);

        // h = LN(attn + h)
        add_ln512_kernel<<<M, 128, 0, stream>>>(t32, h32, g_, bt_, h32, h16);

        // FFN: hid = relu(h @ W1 + b1) -> f16 ; f = hid @ W2 + b2 -> f32
        gemm64_wmma<true, false><<<gF1, 128, 0, stream>>>(
            h16, w1, b1_, nullptr, hid16, 512, 512, 2048, 512, 1, 0, 0, 0, 0, 0, 0);
        gemm64_wmma<false, false><<<gF2, 128, 0, stream>>>(
            hid16, w2, b2_, t32, nullptr, 2048, 2048, 512, 2048, 1, 0, 0, 0, 0, 0, 0);

        // h = LN(ffn + h)   (same gamma/beta reused, per reference)
        add_ln512_kernel<<<M, 128, 0, stream>>>(t32, h32, g_, bt_, h32, h16);
    }

    // final head: [8, 262144] @ [262144, 5] + bout
    out_head_kernel<<<40, 256, 0, stream>>>(h32, Wout, bout, out);
}
